// SwinIRStrip_42649025249318
// MI455X (gfx1250) — compile-verified, matrix operand order checked by reading, not run on hardware
//
#include <hip/hip_runtime.h>
#include <hip/hip_bf16.h>

typedef __attribute__((ext_vector_type(16))) _Float16 v16h;
typedef __attribute__((ext_vector_type(8)))  float    v8f;
typedef __attribute__((ext_vector_type(4)))  int      v4i;

#define BATCH   512
#define NTOK    256
#define NH      6
#define HD      32
#define DIM     192
#define SCALE_F 0.17677669529663687f   // 32^-0.5
#define SROW    260                    // LDS score row stride (floats): 16B aligned rows

#define QKV_W_ELEMS  (576 * 192)
#define PROJ_W_ELEMS (192 * 192)
#define S_BYTES      (NTOK * SROW * 4)                        // 266240
#define KV_BYTES     (2 * NTOK * HD * 2)                      // 32768
#define ATTN_LDS     (S_BYTES + KV_BYTES + NTOK * 4)          // + inv_sum = 300032
#define QKV_LDS      (QKV_W_ELEMS * 2)                        // 221184
#define PROJ_LDS     (PROJ_W_ELEMS * 2)                       // 73728

// -------------------- async staging into LDS (gfx1250 path) ----------------
#if defined(__has_builtin)
#if __has_builtin(__builtin_amdgcn_global_load_async_to_lds_b128)
#define HAVE_ASYNC_LDS 1
#endif
#endif
#ifndef HAVE_ASYNC_LDS
#define HAVE_ASYNC_LDS 0
#endif

__device__ __forceinline__ void stage_lds_16(void* dst_lds, const void* src_global,
                                             int nbytes, int tid, int nthreads) {
    for (int off = tid * 16; off < nbytes; off += nthreads * 16) {
#if HAVE_ASYNC_LDS
        __builtin_amdgcn_global_load_async_to_lds_b128(
            (v4i*)((const char*)src_global + off),
            (v4i*)((char*)dst_lds + off), 0, 0);
#else
        *(int4*)((char*)dst_lds + off) = *(const int4*)((const char*)src_global + off);
#endif
    }
}

__device__ __forceinline__ void stage_wait() {
#if HAVE_ASYNC_LDS
#if __has_builtin(__builtin_amdgcn_s_wait_asynccnt)
    __builtin_amdgcn_s_wait_asynccnt(0);
#else
    asm volatile("s_wait_asynccnt 0" ::: "memory");
#endif
#endif
    __syncthreads();
}

// ---------------------------------------------------------------------------
// 16x32 fragment in CDNA5 WMMA A-layout (ISA 7.12.2, 16-bit A 16x32):
//   lanes 0-15 : row M=lane,  halves[0..7]=K 0..7,  halves[8..15]=K 16..23
//   lanes 16-31: row M=l-16,  halves[0..7]=K 8..15, halves[8..15]=K 24..31
// Same pattern on a transposed source yields the B fragment.
// ---------------------------------------------------------------------------
template <typename T>
__device__ __forceinline__ v16h load_frag(const T* p, int ld, int lane) {
    const int r  = lane & 15;
    const int kb = (lane >> 4) << 3;
    const T* row = p + (size_t)r * ld + kb;
    v16h f;
#pragma unroll
    for (int i = 0; i < 8; ++i) f[i]     = (_Float16)row[i];
#pragma unroll
    for (int i = 0; i < 8; ++i) f[i + 8] = (_Float16)row[16 + i];
    return f;
}

// ===========================================================================
// Kernel 0: one-time weight conversion f32 -> f16 into workspace
// ===========================================================================
__global__ __launch_bounds__(256) void cvt_w_kernel(
    const float* __restrict__ qw, const float* __restrict__ pw,
    _Float16* __restrict__ qwf, _Float16* __restrict__ pwf)
{
    const int i = blockIdx.x * 256 + threadIdx.x;
    if (i < QKV_W_ELEMS)  qwf[i] = (_Float16)qw[i];
    if (i < PROJ_W_ELEMS) pwf[i] = (_Float16)pw[i];
}

// ===========================================================================
// Kernel 1: qkv = x @ qkv_w.T + qkv_b   (M=131072, N=576, K=192)
//   Weights staged once per block into LDS (async); shared by all 8 waves.
//   q -> qf (b,h,n,d) f16 pre-scaled; k -> kf (b,h,n,d); v -> vt (b,h,d,n)
// ===========================================================================
__global__ __launch_bounds__(256) void qkv_kernel(
    const float* __restrict__ x, const _Float16* __restrict__ wf,
    const float* __restrict__ bias,
    _Float16* __restrict__ qf, _Float16* __restrict__ kf,
    _Float16* __restrict__ vt)
{
    extern __shared__ _Float16 Wl[];                 // [576][192] f16
    stage_lds_16(Wl, wf, QKV_LDS, threadIdx.x, 256); // async, overlaps A loads

    const int lane  = threadIdx.x & 31;
    const int wave  = threadIdx.x >> 5;
    const int m0    = (blockIdx.x * 8 + wave) * 16;
    const int nloc  = lane & 15;
    const int mbase = (lane >> 4) << 3;
    const int b     = m0 >> 8;         // 16-row tile never crosses a batch
    const int nbase = (m0 & 255) + mbase;

    v16h a[6];
#pragma unroll
    for (int kt = 0; kt < 6; ++kt)
        a[kt] = load_frag(x + (size_t)m0 * DIM + kt * 32, DIM, lane);

    stage_wait();

    for (int nt = 0; nt < 36; ++nt) {
        v8f c = {};
#pragma unroll
        for (int kt = 0; kt < 6; ++kt) {
            v16h bf = load_frag(Wl + (nt * 16) * DIM + kt * 32, DIM, lane);
            c = __builtin_amdgcn_wmma_f32_16x16x32_f16(
                    false, a[kt], false, bf, (short)0, c, false, false);
        }
        const int   C     = nt * 16 + nloc;        // output channel 0..575
        const float bb    = bias[C];
        const int   which = C / DIM;               // uniform per nt (tiles aligned)
        const int   rem   = C - which * DIM;
        const int   h     = rem >> 5;
        const int   d     = rem & 31;
        const size_t hq   = (size_t)(b * NH + h) << 13;   // *(256*32)
        if (which < 2) {
            const float scale = (which == 0) ? SCALE_F : 1.0f;
            _Float16* dst = ((which == 0) ? qf : kf) + hq + (nbase << 5) + d;
#pragma unroll
            for (int vv = 0; vv < 8; ++vv)
                dst[vv << 5] = (_Float16)((c[vv] + bb) * scale);
        } else {
            _Float16* dst = vt + hq + (d << 8) + nbase;
#pragma unroll
            for (int vv = 0; vv < 8; ++vv)
                dst[vv] = (_Float16)(c[vv] + bb);
        }
    }
}

// ===========================================================================
// Kernel 2: per (batch, head) attention.  k and v^T staged into LDS (async).
//   S = (q*scale) k^T + bias -> LDS f32
//   row max + exp (unnormalized, float4-vectorized), inv_sum -> LDS
//   out = (e @ v) * inv_sum -> of f16   (normalization folded into epilogue)
// ===========================================================================
__global__ __launch_bounds__(256) void attn_kernel(
    const _Float16* __restrict__ qf, const _Float16* __restrict__ kf,
    const _Float16* __restrict__ vt, const float* __restrict__ rpb,
    _Float16* __restrict__ of)
{
    extern __shared__ char smem[];
    float*    S    = (float*)smem;                           // [256][SROW]
    _Float16* kl   = (_Float16*)(smem + S_BYTES);            // [256][32]
    _Float16* vl   = kl + NTOK * HD;                         // [32][256] (v^T)
    float*    invS = (float*)(smem + S_BYTES + KV_BYTES);    // [256]

    const int lane  = threadIdx.x & 31;
    const int wave  = threadIdx.x >> 5;
    const int bh    = blockIdx.x;
    const int b     = bh / NH;
    const int h     = bh - b * NH;
    const _Float16* q = qf + (size_t)bh * NTOK * HD;
    const int nloc  = lane & 15;
    const int mbase = (lane >> 4) << 3;

    stage_lds_16(kl, kf + (size_t)bh * NTOK * HD, NTOK * HD * 2, threadIdx.x, 256);
    stage_lds_16(vl, vt + (size_t)bh * NTOK * HD, NTOK * HD * 2, threadIdx.x, 256);

    // ---- Phase 1: scores (K=32 -> single WMMA per 16x16 tile)
    for (int mi = 0; mi < 2; ++mi) {
        const int mt = wave * 2 + mi;
        v16h aq = load_frag(q + (size_t)mt * 16 * HD, HD, lane);   // global, per-wave
        if (mi == 0) stage_wait();                                  // k/v now in LDS
        for (int nt = 0; nt < 16; ++nt) {
            v16h bk = load_frag(kl + nt * 16 * HD, HD, lane);
            v8f c = {};
            c = __builtin_amdgcn_wmma_f32_16x16x32_f16(
                    false, aq, false, bk, (short)0, c, false, false);
            const int n  = nt * 16 + nloc;
            const int yn = n >> 3, xn = n & 7;                      // SW = 8
#pragma unroll
            for (int vv = 0; vv < 8; ++vv) {
                const int m   = mt * 16 + mbase + vv;
                const int ym  = m >> 3, xm = m & 7;
                const int idx = (ym - yn + 31) * 15 + (xm - xn + 7);
                S[m * SROW + n] = c[vv] + rpb[idx * NH + h];
            }
        }
    }
    __syncthreads();

    // ---- Phase 2: row max + exp (unnormalized), one row per thread, float4
    {
        float4* row4 = (float4*)(S + (size_t)threadIdx.x * SROW);
        float4 m4 = row4[0];
        for (int j = 1; j < 64; ++j) {
            float4 t = row4[j];
            m4.x = fmaxf(m4.x, t.x); m4.y = fmaxf(m4.y, t.y);
            m4.z = fmaxf(m4.z, t.z); m4.w = fmaxf(m4.w, t.w);
        }
        const float mx = fmaxf(fmaxf(m4.x, m4.y), fmaxf(m4.z, m4.w));
        float sum = 0.f;
        for (int j = 0; j < 64; ++j) {
            float4 t = row4[j];
            t.x = __expf(t.x - mx); t.y = __expf(t.y - mx);
            t.z = __expf(t.z - mx); t.w = __expf(t.w - mx);
            row4[j] = t;
            sum += (t.x + t.y) + (t.z + t.w);
        }
        invS[threadIdx.x] = 1.f / sum;
    }
    __syncthreads();

    // ---- Phase 3: out = (e @ v) * inv_sum.  A from LDS f32, B from LDS v^T.
    for (int mi = 0; mi < 2; ++mi) {
        const int mt = wave * 2 + mi;
#pragma unroll
        for (int dt = 0; dt < 2; ++dt) {
            v8f c = {};
#pragma unroll
            for (int kt = 0; kt < 8; ++kt) {
                v16h ap = load_frag(S + (size_t)mt * 16 * SROW + kt * 32, SROW, lane);
                v16h bv = load_frag(vl + (dt * 16) * NTOK + kt * 32, NTOK, lane);
                c = __builtin_amdgcn_wmma_f32_16x16x32_f16(
                        false, ap, false, bv, (short)0, c, false, false);
            }
            const int d = dt * 16 + nloc;
            _Float16* dst = of + (size_t)(b * NTOK + mt * 16 + mbase) * DIM + h * HD + d;
            const float* is = invS + mt * 16 + mbase;
#pragma unroll
            for (int vv = 0; vv < 8; ++vv)
                dst[(size_t)vv * DIM] = (_Float16)(c[vv] * is[vv]);
        }
    }
}

// ===========================================================================
// Kernel 3: out = o @ proj_w.T + proj_b   (M=131072, N=192, K=192), fp32 out
//   proj_w f16 staged into LDS once per block (async).
// ===========================================================================
__global__ __launch_bounds__(256) void proj_kernel(
    const _Float16* __restrict__ of, const _Float16* __restrict__ wf,
    const float* __restrict__ bias, float* __restrict__ out)
{
    extern __shared__ _Float16 Wl[];                  // [192][192] f16
    stage_lds_16(Wl, wf, PROJ_LDS, threadIdx.x, 256);

    const int lane  = threadIdx.x & 31;
    const int wave  = threadIdx.x >> 5;
    const int m0    = (blockIdx.x * 8 + wave) * 16;
    const int nloc  = lane & 15;
    const int mbase = (lane >> 4) << 3;

    v16h a[6];
#pragma unroll
    for (int kt = 0; kt < 6; ++kt)
        a[kt] = load_frag(of + (size_t)m0 * DIM + kt * 32, DIM, lane);

    stage_wait();

    for (int nt = 0; nt < 12; ++nt) {
        v8f c = {};
#pragma unroll
        for (int kt = 0; kt < 6; ++kt) {
            v16h bf = load_frag(Wl + (nt * 16) * DIM + kt * 32, DIM, lane);
            c = __builtin_amdgcn_wmma_f32_16x16x32_f16(
                    false, a[kt], false, bf, (short)0, c, false, false);
        }
        const int   C  = nt * 16 + nloc;
        const float bb = bias[C];
        float* dst = out + (size_t)(m0 + mbase) * DIM + C;
#pragma unroll
        for (int vv = 0; vv < 8; ++vv)
            dst[(size_t)vv * DIM] = c[vv] + bb;
    }
}

// ===========================================================================
extern "C" void kernel_launch(void* const* d_in, const int* in_sizes, int n_in,
                              void* d_out, int out_size, void* d_ws, size_t ws_size,
                              hipStream_t stream)
{
    const float* x      = (const float*)d_in[0];
    const float* qkv_w  = (const float*)d_in[1];
    const float* qkv_b  = (const float*)d_in[2];
    const float* proj_w = (const float*)d_in[3];
    const float* proj_b = (const float*)d_in[4];
    const float* rpb    = (const float*)d_in[5];
    float*       out    = (float*)d_out;

    const size_t per = (size_t)BATCH * NH * NTOK * HD;    // 25.2M f16 each
    _Float16* qf  = (_Float16*)d_ws;
    _Float16* kf  = qf  + per;
    _Float16* vt  = kf  + per;
    _Float16* of  = vt  + per;                            // (b, n, 192) f16
    _Float16* qwf = of  + per;                            // f16 weights
    _Float16* pwf = qwf + QKV_W_ELEMS;

    (void)hipFuncSetAttribute((const void*)qkv_kernel,
                              hipFuncAttributeMaxDynamicSharedMemorySize, QKV_LDS);
    (void)hipFuncSetAttribute((const void*)attn_kernel,
                              hipFuncAttributeMaxDynamicSharedMemorySize, ATTN_LDS);
    (void)hipFuncSetAttribute((const void*)proj_kernel,
                              hipFuncAttributeMaxDynamicSharedMemorySize, PROJ_LDS);

    cvt_w_kernel<<<(QKV_W_ELEMS + 255) / 256, 256, 0, stream>>>(qkv_w, proj_w, qwf, pwf);
    qkv_kernel  <<<1024,       256, QKV_LDS,  stream>>>(x, qwf, qkv_b, qf, kf, vt);
    attn_kernel <<<BATCH * NH, 256, ATTN_LDS, stream>>>(qf, kf, vt, rpb, of);
    proj_kernel <<<1024,       256, PROJ_LDS, stream>>>(of, pwf, proj_b, out);
}
